// Relation_82042465288549
// MI455X (gfx1250) — matrix operand order repaired and checked by reading, not compiled, not run on hardware
//
#include <hip/hip_runtime.h>

typedef __attribute__((ext_vector_type(16))) __bf16 v16bf;
typedef __attribute__((ext_vector_type(8)))  __bf16 v8bf;
typedef __attribute__((ext_vector_type(8)))  float  v8f;

#define D_TOT 768
#define C_IN  256
#define NPIX  4096
#define NBATCH 4

__device__ __forceinline__ unsigned f32_to_bf16_bits(float f) {
  union { float f; unsigned u; } v; v.f = f;
  unsigned u = v.u;
  u += 0x7FFFu + ((u >> 16) & 1u);   // round-to-nearest-even
  return u >> 16;
}

__device__ __forceinline__ unsigned pack_bf16x2(float lo, float hi) {
  return f32_to_bf16_bits(lo) | (f32_to_bf16_bits(hi) << 16);
}

__device__ __forceinline__ v16bf mk16(v8bf lo, v8bf hi) {
  v16bf r;
#pragma unroll
  for (int i = 0; i < 8; ++i) { r[i] = lo[i]; r[i + 8] = hi[i]; }
  return r;
}

// ---------------------------------------------------------------------------
// Phase 1: qkv[n][d][p] = sum_c w[d][c] * x[n][c][p] + bias[d]
// 64x64 output tile / block, K-slab = 32 (one WMMA K), bf16 WMMA accum f32.
// LDS staging packs two bf16 per u32 -> ds_store_b32 (half the DS ops).
// ---------------------------------------------------------------------------
__global__ __launch_bounds__(256) void qkv_gemm_wmma(
    const float* __restrict__ x, const float* __restrict__ w,
    const float* __restrict__ bias, float* __restrict__ qkv) {
  __shared__ __align__(16) unsigned As32[64 * 16];  // [d][k/2] packed bf16x2
  __shared__ __align__(16) unsigned Bt32[64 * 16];  // [p][k/2] packed bf16x2

  const int t     = threadIdx.x;
  const int lane  = t & 31;
  const int wave  = t >> 5;
  const int lhalf = lane >> 4;     // 0: lanes 0-15, 1: lanes 16-31
  const int lm    = lane & 15;

  const int tileD0 = blockIdx.y * 64;
  const int tileP0 = blockIdx.x * 64;
  const int n      = blockIdx.z;
  const float* xn  = x + (size_t)n * C_IN * NPIX;

  const int m  = (2 * wave) >> 2;  // M-subtile (0..3), shared by the pair
  const int n0 = (2 * wave) & 3;   // N-subtiles n0 and n0+1

  v8f acc0 = {}; v8f acc1 = {};

  for (int k0 = 0; k0 < C_IN; k0 += 32) {
#pragma unroll
    for (int e = 0; e < 4; ++e) {
      int pl = t + e * 256;                  // 0..1023 (u32 pairs)
      // A tile: row ar (d), k-pair ac; one 8-byte global load, one b32 DS store.
      int ar = pl >> 4, ac = (pl & 15) * 2;
      float2 wv = *(const float2*)&w[(size_t)(tileD0 + ar) * C_IN + k0 + ac];
      As32[pl] = pack_bf16x2(wv.x, wv.y);
      // B tile (transposed): pixel bp, k-pair bc2; two lane-coalesced loads.
      int bp = pl & 63, bc2 = pl >> 6;
      size_t gb = (size_t)(k0 + bc2 * 2) * NPIX + tileP0 + bp;
      Bt32[bp * 16 + bc2] = pack_bf16x2(xn[gb], xn[gb + NPIX]);
    }
    __syncthreads();

    // A frag: lane holds M = m*16+lm; K chunks {2*lhalf, 2*lhalf+1} of 8 bf16.
    const v8bf* Arow = (const v8bf*)(&As32[(m * 16 + lm) * 16]);
    v16bf afrag = mk16(Arow[2 * lhalf], Arow[2 * lhalf + 1]);
    // B frags: lane holds N = n*16+lm; K chunks {2*lhalf, 2*lhalf+1}.
    const v8bf* Brow0 = (const v8bf*)(&Bt32[((n0    ) * 16 + lm) * 16]);
    const v8bf* Brow1 = (const v8bf*)(&Bt32[((n0 + 1) * 16 + lm) * 16]);
    v16bf bfrag0 = mk16(Brow0[2 * lhalf], Brow0[2 * lhalf + 1]);
    v16bf bfrag1 = mk16(Brow1[2 * lhalf], Brow1[2 * lhalf + 1]);

    acc0 = __builtin_amdgcn_wmma_f32_16x16x32_bf16(
        false, afrag, false, bfrag0, (short)0, acc0, false, false);
    acc1 = __builtin_amdgcn_wmma_f32_16x16x32_bf16(
        false, afrag, false, bfrag1, (short)0, acc1, false, false);
    __syncthreads();
  }

  const size_t obase = (size_t)n * D_TOT * NPIX;
  const int p0 = tileP0 + n0 * 16 + lm;
  const int p1 = p0 + 16;
#pragma unroll
  for (int r = 0; r < 8; ++r) {
    int d = tileD0 + m * 16 + r + (lhalf ? 8 : 0);   // C/D layout: VGPR r -> M
    float bv = bias[d];
    qkv[obase + (size_t)d * NPIX + p0] = acc0[r] + bv;
    qkv[obase + (size_t)d * NPIX + p1] = acc1[r] + bv;
  }
}

// ---------------------------------------------------------------------------
// Phase 2: 5x5 neighborhood attention, thread-per-(n, head, pixel).
// qkv is L2-resident (50 MB < 192 MB L2); loads coalesced across lanes (pixels).
// ---------------------------------------------------------------------------
__global__ __launch_bounds__(256) void natt_kernel(
    const float* __restrict__ qkv, float* __restrict__ out) {
  int tid = blockIdx.x * 256 + threadIdx.x;   // n*8*4096 threads
  int pix = tid & 4095;
  int g   = (tid >> 12) & 7;
  int n   = tid >> 15;
  int y = pix >> 6, xc = pix & 63;

  const float* Qp = qkv + ((size_t)n * D_TOT + g * 32) * NPIX;
  const float* Kp = Qp + (size_t)256 * NPIX;
  const float* Vp = Qp + (size_t)512 * NPIX;

  float sc[25];
  int   nb[25];
  unsigned vm = 0;
#pragma unroll
  for (int k = 0; k < 25; ++k) {
    int yy = y + (k / 5) - 2, xx = xc + (k % 5) - 2;
    bool v = (yy >= 0) & (yy < 64) & (xx >= 0) & (xx < 64);
    vm |= (v ? (1u << k) : 0u);
    nb[k] = v ? (yy * 64 + xx) : pix;   // clamp invalid to a safe address
    sc[k] = 0.f;
  }
#pragma unroll 4
  for (int d = 0; d < 32; ++d) {
    float qd = Qp[(size_t)d * NPIX + pix];
    const float* Kd = Kp + (size_t)d * NPIX;
#pragma unroll
    for (int k = 0; k < 25; ++k) sc[k] = fmaf(qd, Kd[nb[k]], sc[k]);
  }
  float mx = -3.402823466e38f;
#pragma unroll
  for (int k = 0; k < 25; ++k)
    if ((vm >> k) & 1) mx = fmaxf(mx, sc[k]);
  float ssum = 0.f;
#pragma unroll
  for (int k = 0; k < 25; ++k) {
    float e = ((vm >> k) & 1) ? __expf(sc[k] - mx) : 0.f;
    sc[k] = e; ssum += e;
  }
  float inv = 1.f / ssum;
#pragma unroll 4
  for (int d = 0; d < 32; ++d) {
    const float* Vd = Vp + (size_t)d * NPIX;
    float acc = 0.f;
#pragma unroll
    for (int k = 0; k < 25; ++k) acc = fmaf(sc[k], Vd[nb[k]], acc);
    out[((size_t)n * 256 + g * 32 + d) * NPIX + pix] = acc * inv;
  }
}

extern "C" void kernel_launch(void* const* d_in, const int* in_sizes, int n_in,
                              void* d_out, int out_size, void* d_ws, size_t ws_size,
                              hipStream_t stream) {
  (void)in_sizes; (void)n_in; (void)out_size; (void)ws_size;
  const float* x = (const float*)d_in[0];   // (4, 256, 64, 64)
  const float* w = (const float*)d_in[1];   // (768, 256)
  const float* b = (const float*)d_in[2];   // (768,)
  float* out = (float*)d_out;               // (4, 256, 64, 64)
  float* qkv = (float*)d_ws;                // 4*768*4096 f32 = 50.3 MB scratch

  dim3 g1(NPIX / 64, D_TOT / 64, NBATCH);   // (64, 12, 4)
  qkv_gemm_wmma<<<g1, 256, 0, stream>>>(x, w, b, qkv);

  int total = NBATCH * 8 * NPIX;            // 131072 threads
  natt_kernel<<<total / 256, 256, 0, stream>>>(qkv, out);
}
